// CFDError_19645180412070
// MI455X (gfx1250) — matrix operand order repaired and checked by reading, not compiled, not run on hardware
//
#include <hip/hip_runtime.h>
#include <hip/hip_bf16.h>
#include <math.h>

typedef __attribute__((ext_vector_type(16))) _Float16 v16h;
typedef __attribute__((ext_vector_type(8)))  float    v8f;
typedef __attribute__((ext_vector_type(4)))  int      v4i;

#define BN_EPS 1e-5f

#if __has_builtin(__builtin_amdgcn_global_load_async_to_lds_b128) && \
    __has_builtin(__builtin_amdgcn_s_wait_asynccnt)
#define USE_ASYNC_LDS 1
#else
#define USE_ASYNC_LDS 0
#endif

#define ASG __attribute__((address_space(1)))
#define ASL __attribute__((address_space(3)))

// Hardware tanh (gfx1250 v_tanh_f32) with safe fallbacks.
__device__ __forceinline__ float fast_tanh(float x) {
#if __has_builtin(__builtin_amdgcn_tanhf)
  return __builtin_amdgcn_tanhf(x);
#elif __has_builtin(__builtin_amdgcn_tanh_f32)
  return __builtin_amdgcn_tanh_f32(x);
#else
  float t = __expf(-2.0f * __builtin_fabsf(x));
  float y = (1.0f - t) / (1.0f + t);
  return __builtin_copysignf(y, x);
#endif
}

// K index held by element i (0..15) of a 16-bit A/B fragment for lane-half hs
// (ISA 7.12.2: VGPR0..3 hold K 0..7 / 8..15, VGPR4..7 hold K 16..23 / 24..31)
__device__ __forceinline__ int kmap16(int hs, int i) {
  return (i < 8) ? (hs * 8 + i) : (16 + hs * 8 + (i - 8));
}

// Copy one 16B chunk global->LDS (async-to-LDS when available).
__device__ __forceinline__ void copy16(_Float16* ldst, const _Float16* gsrc) {
#if USE_ASYNC_LDS
  __builtin_amdgcn_global_load_async_to_lds_b128(
      (ASG v4i*)gsrc, (ASL v4i*)ldst, 0, 0);
#else
  *(float4*)ldst = *(const float4*)gsrc;
#endif
}
__device__ __forceinline__ void copy_fence() {
#if USE_ASYNC_LDS
  __builtin_amdgcn_s_wait_asynccnt(0);
#endif
}

// ---------------------------------------------------------------------------
// Stage 1: gather t=[x_i, x_j-x_i] -> LDS(f16), WMMA (KPAD x 32), +b1, tanh,
// store h (f16, [E,32]), accumulate BN column sum/sumsq.
// ---------------------------------------------------------------------------
template<int CIN, int KPAD, int TILE>
__global__ __launch_bounds__(TILE * 2)
void edge_stage1(const float* __restrict__ x,
                 const long long* __restrict__ src,
                 const long long* __restrict__ dst,
                 const float* __restrict__ w1,
                 const float* __restrict__ b1,
                 _Float16* __restrict__ h_out,
                 float* __restrict__ stats,   // [sum32, sumsq32, scale32, shift32]
                 int E) {
  __shared__ alignas(16) _Float16 t_lds[TILE][KPAD];
  __shared__ alignas(16) _Float16 w_lds[KPAD][32];
  __shared__ float s_red[64];

  const int tid  = threadIdx.x;
  const int nthr = TILE * 2;
  const int ebase = blockIdx.x * TILE;
  const bool full = (ebase + TILE) <= E;

  for (int i = tid; i < KPAD * 32; i += nthr) {
    int r = i >> 5, c = i & 31;
    w_lds[r][c] = (_Float16)((r < 2 * CIN) ? w1[r * 32 + c] : 0.0f);
  }
  if (tid < 64) s_red[tid] = 0.0f;

  if (full) {
    for (int i = tid; i < TILE * KPAD; i += nthr) {
      int el = i / KPAD, k = i % KPAD;
      int e = ebase + el;
      float v = 0.0f;
      int di = (int)dst[e];
      if (k < CIN) {
        v = x[di * CIN + k];
      } else if (k < 2 * CIN) {
        int si = (int)src[e];
        int kk = k - CIN;
        v = x[si * CIN + kk] - x[di * CIN + kk];
      }
      t_lds[el][k] = (_Float16)v;
    }
  } else {
    for (int i = tid; i < TILE * KPAD; i += nthr) {
      int el = i / KPAD, k = i % KPAD;
      int e = ebase + el;
      float v = 0.0f;
      if (e < E) {
        int di = (int)dst[e];
        if (k < CIN) {
          v = x[di * CIN + k];
        } else if (k < 2 * CIN) {
          int si = (int)src[e];
          int kk = k - CIN;
          v = x[si * CIN + kk] - x[di * CIN + kk];
        }
      }
      t_lds[el][k] = (_Float16)v;
    }
  }
  __syncthreads();

  const int lane = tid & 31;
  const int wave = tid >> 5;
  const int hs   = lane >> 4;
  const int nrow = lane & 15;
  const int erow = wave * 16 + nrow;

  v8f acc0 = {}; v8f acc1 = {};
#pragma unroll
  for (int ks = 0; ks < KPAD / 32; ++ks) {
    const int kb = ks * 32;
    v16h a, b0, b1v;
#pragma unroll
    for (int i = 0; i < 16; ++i) a[i] = t_lds[erow][kb + kmap16(hs, i)];
    const int krow = kb + nrow + 16 * hs;
#pragma unroll
    for (int i = 0; i < 16; ++i) { b0[i] = w_lds[kb + kmap16(hs, i)][nrow]; b1v[i] = w_lds[kb + kmap16(hs, i)][16 + nrow]; }
    acc0 = __builtin_amdgcn_wmma_f32_16x16x32_f16(false, a, false, b0,  (short)0, acc0, false, false);
    acc1 = __builtin_amdgcn_wmma_f32_16x16x32_f16(false, a, false, b1v, (short)0, acc1, false, false);
  }

  if (full) {
    // branch-free epilogue: tanh -> wave-owned rows of t_lds -> b128 stores
#pragma unroll
    for (int nt = 0; nt < 2; ++nt) {
      const v8f acc = nt ? acc1 : acc0;
      const int col = nt * 16 + nrow;
      const float bias = b1[col];
      float s = 0.f, s2 = 0.f;
#pragma unroll
      for (int r = 0; r < 8; ++r) {
        float hv = fast_tanh(acc[r] + bias);
        t_lds[wave * 16 + r + 8 * hs][col] = (_Float16)hv;
        s += hv; s2 += hv * hv;
      }
      atomicAdd(&s_red[col], s);
      atomicAdd(&s_red[32 + col], s2);
    }
#pragma unroll
    for (int c2 = 0; c2 < 2; ++c2) {
      int j = lane + c2 * 32;            // 64 chunks of 16B per wave
      int row = wave * 16 + (j >> 2);
      int off = (j & 3) * 8;
      *(float4*)&h_out[(size_t)(ebase + row) * 32 + off] = *(const float4*)&t_lds[row][off];
    }
  } else {
#pragma unroll
    for (int nt = 0; nt < 2; ++nt) {
      const v8f acc = nt ? acc1 : acc0;
      const int col = nt * 16 + nrow;
      const float bias = b1[col];
      float s = 0.f, s2 = 0.f;
#pragma unroll
      for (int r = 0; r < 8; ++r) {
        int e = ebase + wave * 16 + r + 8 * hs;
        if (e < E) {
          float hv = fast_tanh(acc[r] + bias);
          h_out[e * 32 + col] = (_Float16)hv;
          s += hv; s2 += hv * hv;
        }
      }
      atomicAdd(&s_red[col], s);
      atomicAdd(&s_red[32 + col], s2);
    }
  }
  __syncthreads();
  if (tid < 64) atomicAdd(&stats[tid], s_red[tid]);
}

// ---------------------------------------------------------------------------
// BN finalize: fold mean/var + gamma/beta into scale/shift.
// ---------------------------------------------------------------------------
__global__ void bn_finalize(float* __restrict__ stats,
                            const float* __restrict__ g,
                            const float* __restrict__ be,
                            float invE) {
  int c = threadIdx.x;
  if (c < 32) {
    float mu  = stats[c] * invE;
    float var = stats[32 + c] * invE - mu * mu;
    float rs  = rsqrtf(var + BN_EPS);
    float sc  = rs * g[c];
    stats[64 + c] = sc;
    stats[96 + c] = be[c] - mu * sc;
  }
}

// ---------------------------------------------------------------------------
// Stage 2: raw h tile -> LDS (async b128), BN affine folded into A-fragment
// build, WMMA (32x32), +b2, tanh -> h in place; BN2 sums.
// ---------------------------------------------------------------------------
__global__ __launch_bounds__(128)
void edge_stage2(_Float16* __restrict__ h,
                 const float* __restrict__ stats1,  // scale @64, shift @96
                 const float* __restrict__ w2,
                 const float* __restrict__ b2,
                 float* __restrict__ stats2,
                 int E) {
  __shared__ alignas(16) _Float16 t_lds[64][32];
  __shared__ alignas(16) _Float16 w_lds[32][32];
  __shared__ float s_red[64];

  const int tid = threadIdx.x;
  const int ebase = blockIdx.x * 64;
  const bool full = (ebase + 64) <= E;

  for (int i = tid; i < 32 * 32; i += 128)
    w_lds[i >> 5][i & 31] = (_Float16)w2[i];
  if (tid < 64) s_red[tid] = 0.0f;

  {
    const _Float16* gsrc = h + (size_t)ebase * 32;
    _Float16* ldst = &t_lds[0][0];
    if (full) {
#pragma unroll
      for (int it = 0; it < 2; ++it) {
        int i = tid + it * 128;
        copy16(ldst + i * 8, gsrc + i * 8);
      }
    } else {
#pragma unroll
      for (int it = 0; it < 2; ++it) {
        int i = tid + it * 128;          // 16B chunk index; edge = i/4
        if (ebase + (i >> 2) < E) copy16(ldst + i * 8, gsrc + i * 8);
      }
    }
  }
  copy_fence();
  __syncthreads();

  const int lane = tid & 31;
  const int wave = tid >> 5;
  const int hs   = lane >> 4;
  const int nrow = lane & 15;
  const int erow = wave * 16 + nrow;

  // per-lane BN coefficients for the 16 static K indices
  float sc[16], sh[16];
#pragma unroll
  for (int i = 0; i < 16; ++i) {
    int k = kmap16(hs, i);
    sc[i] = stats1[64 + k];
    sh[i] = stats1[96 + k];
  }

  v16h a, b0, b1v;
#pragma unroll
  for (int i = 0; i < 16; ++i) {
    float v = (float)t_lds[erow][kmap16(hs, i)] * sc[i] + sh[i];
    a[i] = (_Float16)v;
  }
  const int krow = nrow + 16 * hs;
#pragma unroll
  for (int i = 0; i < 16; ++i) { b0[i] = w_lds[kmap16(hs, i)][nrow]; b1v[i] = w_lds[kmap16(hs, i)][16 + nrow]; }
  v8f acc0 = {}; v8f acc1 = {};
  acc0 = __builtin_amdgcn_wmma_f32_16x16x32_f16(false, a, false, b0,  (short)0, acc0, false, false);
  acc1 = __builtin_amdgcn_wmma_f32_16x16x32_f16(false, a, false, b1v, (short)0, acc1, false, false);

  if (full) {
#pragma unroll
    for (int nt = 0; nt < 2; ++nt) {
      const v8f acc = nt ? acc1 : acc0;
      const int col = nt * 16 + nrow;
      const float bias = b2[col];
      float s = 0.f, s2 = 0.f;
#pragma unroll
      for (int r = 0; r < 8; ++r) {
        float hv = fast_tanh(acc[r] + bias);
        t_lds[wave * 16 + r + 8 * hs][col] = (_Float16)hv;
        s += hv; s2 += hv * hv;
      }
      atomicAdd(&s_red[col], s);
      atomicAdd(&s_red[32 + col], s2);
    }
#pragma unroll
    for (int c2 = 0; c2 < 2; ++c2) {
      int j = lane + c2 * 32;
      int row = wave * 16 + (j >> 2);
      int off = (j & 3) * 8;
      *(float4*)&h[(size_t)(ebase + row) * 32 + off] = *(const float4*)&t_lds[row][off];
    }
  } else {
#pragma unroll
    for (int nt = 0; nt < 2; ++nt) {
      const v8f acc = nt ? acc1 : acc0;
      const int col = nt * 16 + nrow;
      const float bias = b2[col];
      float s = 0.f, s2 = 0.f;
#pragma unroll
      for (int r = 0; r < 8; ++r) {
        int e = ebase + wave * 16 + r + 8 * hs;
        if (e < E) {
          float hv = fast_tanh(acc[r] + bias);
          h[e * 32 + col] = (_Float16)hv;
          s += hv; s2 += hv * hv;
        }
      }
      atomicAdd(&s_red[col], s);
      atomicAdd(&s_red[32 + col], s2);
    }
  }
  __syncthreads();
  if (tid < 64) atomicAdd(&stats2[tid], s_red[tid]);
}

// ---------------------------------------------------------------------------
// Stage 3: raw h tile -> LDS, BN affine in fragment build, WMMA (32 x COUT),
// +b3, scatter-add to y[dst].
// ---------------------------------------------------------------------------
template<int COUT, int COUTP>
__global__ __launch_bounds__(128)
void edge_stage3(const _Float16* __restrict__ h,
                 const float* __restrict__ stats2,
                 const float* __restrict__ w3,
                 const float* __restrict__ b3,
                 const long long* __restrict__ dst,
                 float* __restrict__ y, int ldy, int col0, int E) {
  __shared__ alignas(16) _Float16 t_lds[64][32];
  __shared__ alignas(16) _Float16 w_lds[32][COUTP];
  __shared__ int s_dst[64];

  const int tid = threadIdx.x;
  const int ebase = blockIdx.x * 64;
  const bool full = (ebase + 64) <= E;

  for (int i = tid; i < 32 * COUTP; i += 128) {
    int r = i / COUTP, c = i % COUTP;
    w_lds[r][c] = (_Float16)((c < COUT) ? w3[r * COUT + c] : 0.0f);
  }
  if (tid < 64) {
    int e = ebase + tid;
    s_dst[tid] = (e < E) ? (int)dst[e] : 0;
  }
  {
    const _Float16* gsrc = h + (size_t)ebase * 32;
    _Float16* ldst = &t_lds[0][0];
    if (full) {
#pragma unroll
      for (int it = 0; it < 2; ++it) {
        int i = tid + it * 128;
        copy16(ldst + i * 8, gsrc + i * 8);
      }
    } else {
#pragma unroll
      for (int it = 0; it < 2; ++it) {
        int i = tid + it * 128;
        if (ebase + (i >> 2) < E) copy16(ldst + i * 8, gsrc + i * 8);
      }
    }
  }
  copy_fence();
  __syncthreads();

  const int lane = tid & 31;
  const int wave = tid >> 5;
  const int hs   = lane >> 4;
  const int nrow = lane & 15;
  const int erow = wave * 16 + nrow;
  constexpr int NT = COUTP / 16;

  float sc[16], sh[16];
#pragma unroll
  for (int i = 0; i < 16; ++i) {
    int k = kmap16(hs, i);
    sc[i] = stats2[64 + k];
    sh[i] = stats2[96 + k];
  }

  v16h a;
#pragma unroll
  for (int i = 0; i < 16; ++i) {
    float v = (float)t_lds[erow][kmap16(hs, i)] * sc[i] + sh[i];
    a[i] = (_Float16)v;
  }
  const int krow = nrow + 16 * hs;

  v8f acc[NT];
#pragma unroll
  for (int nt = 0; nt < NT; ++nt) {
    v16h b;
#pragma unroll
    for (int i = 0; i < 16; ++i) b[i] = w_lds[kmap16(hs, i)][nt * 16 + nrow];
    v8f z = {};
    acc[nt] = __builtin_amdgcn_wmma_f32_16x16x32_f16(false, a, false, b, (short)0, z, false, false);
  }

  if (full) {
#pragma unroll
    for (int nt = 0; nt < NT; ++nt) {
      const int col = nt * 16 + nrow;
      if (COUT == COUTP || col < COUT) {
        const float bias = b3[col];
#pragma unroll
        for (int r = 0; r < 8; ++r) {
          int el = wave * 16 + r + 8 * hs;
          atomicAdd(&y[(size_t)s_dst[el] * ldy + col0 + col], acc[nt][r] + bias);
        }
      }
    }
  } else {
#pragma unroll
    for (int nt = 0; nt < NT; ++nt) {
      const int col = nt * 16 + nrow;
      if (COUT == COUTP || col < COUT) {
        const float bias = b3[col];
#pragma unroll
        for (int r = 0; r < 8; ++r) {
          int el = wave * 16 + r + 8 * hs;
          int e = ebase + el;
          if (e < E) {
            atomicAdd(&y[(size_t)s_dst[el] * ldy + col0 + col], acc[nt][r] + bias);
          }
        }
      }
    }
  }
}

// ---------------------------------------------------------------------------
// Elementwise (float4-vectorized): square xc[:,64:128]; xc[:,128:192]=t3^2+t4^2
// ---------------------------------------------------------------------------
__global__ void square_cols(float4* __restrict__ xc4, int n4) {  // n4 = N*16
  int i = blockIdx.x * blockDim.x + threadIdx.x;
  if (i < n4) {
    int n = i >> 4, c = i & 15;
    float4* p = &xc4[n * 48 + 16 + c];
    float4 v = *p;
    v.x *= v.x; v.y *= v.y; v.z *= v.z; v.w *= v.w;
    *p = v;
  }
}

__global__ void combine4(const float4* __restrict__ t3, const float4* __restrict__ t4,
                         float4* __restrict__ xc4, int n4) {     // n4 = N*16
  int i = blockIdx.x * blockDim.x + threadIdx.x;
  if (i < n4) {
    int n = i >> 4, c = i & 15;
    float4 a = t3[i], b = t4[i];
    float4 r;
    r.x = a.x * a.x + b.x * b.x;
    r.y = a.y * a.y + b.y * b.y;
    r.z = a.z * a.z + b.z * b.z;
    r.w = a.w * a.w + b.w * b.w;
    xc4[n * 48 + 32 + c] = r;
  }
}

// ---------------------------------------------------------------------------
// Host-side driver
// ---------------------------------------------------------------------------
struct ConvParams {
  const float *w1, *b1, *g1, *be1, *w2, *b2, *g2, *be2, *w3, *b3;
};

static void run_conv(int cin, const float* x_in,
                     const long long* src, const long long* dst,
                     const ConvParams& p, float* y, int ldy, int col0, int cout,
                     _Float16* h, float* statsA, float* statsB,
                     int E, hipStream_t stream) {
  const int blocks64 = (E + 63) / 64;
  if (cin == 3)
    edge_stage1<3, 32, 64><<<blocks64, 128, 0, stream>>>(x_in, src, dst, p.w1, p.b1, h, statsA, E);
  else if (cin == 64)
    edge_stage1<64, 128, 64><<<blocks64, 128, 0, stream>>>(x_in, src, dst, p.w1, p.b1, h, statsA, E);
  else
    edge_stage1<192, 384, 32><<<(E + 31) / 32, 64, 0, stream>>>(x_in, src, dst, p.w1, p.b1, h, statsA, E);

  bn_finalize<<<1, 32, 0, stream>>>(statsA, p.g1, p.be1, 1.0f / (float)E);
  edge_stage2<<<blocks64, 128, 0, stream>>>(h, statsA, p.w2, p.b2, statsB, E);
  bn_finalize<<<1, 32, 0, stream>>>(statsB, p.g2, p.be2, 1.0f / (float)E);

  if (cout == 64)
    edge_stage3<64, 64><<<blocks64, 128, 0, stream>>>(h, statsB, p.w3, p.b3, dst, y, ldy, col0, E);
  else
    edge_stage3<3, 16><<<blocks64, 128, 0, stream>>>(h, statsB, p.w3, p.b3, dst, y, ldy, col0, E);
}

extern "C" void kernel_launch(void* const* d_in, const int* in_sizes, int n_in,
                              void* d_out, int out_size, void* d_ws, size_t ws_size,
                              hipStream_t stream) {
  (void)n_in; (void)ws_size;
  const float*     pos  = (const float*)d_in[0];
  const long long* edge = (const long long*)d_in[1];   // int64 edge_index [2,E]
  const int N = in_sizes[0] / 3;
  const int E = in_sizes[1] / 2;
  const long long* src = edge;
  const long long* dst = edge + E;

  ConvParams P[6];
  for (int l = 0; l < 6; ++l) {
    const int b = 2 + l * 10;   // dict insertion order: w1,b1,g1,be1,w2,b2,g2,be2,w3,b3
    P[l].w1  = (const float*)d_in[b + 0];
    P[l].b1  = (const float*)d_in[b + 1];
    P[l].g1  = (const float*)d_in[b + 2];
    P[l].be1 = (const float*)d_in[b + 3];
    P[l].w2  = (const float*)d_in[b + 4];
    P[l].b2  = (const float*)d_in[b + 5];
    P[l].g2  = (const float*)d_in[b + 6];
    P[l].be2 = (const float*)d_in[b + 7];
    P[l].w3  = (const float*)d_in[b + 8];
    P[l].b3  = (const float*)d_in[b + 9];
  }

  // workspace carve-out (256B aligned)
  char* ws = (char*)d_ws;
  size_t off = 0;
  auto carve = [&](size_t bytes) -> void* {
    void* p = ws + off;
    off = (off + bytes + 255) & ~(size_t)255;
    return p;
  };
  _Float16* h   = (_Float16*)carve((size_t)E * 32 * sizeof(_Float16));
  float* xbuf   = (float*)carve((size_t)N * 64  * sizeof(float));
  float* xc     = (float*)carve((size_t)N * 192 * sizeof(float));
  float* t3     = (float*)carve((size_t)N * 64  * sizeof(float));
  float* t4     = (float*)carve((size_t)N * 64  * sizeof(float));
  float* stats  = (float*)carve(12 * 128 * sizeof(float));

  // zero all accumulators every call (deterministic under graph replay)
  (void)hipMemsetAsync(xbuf,  0, (size_t)N * 64  * sizeof(float), stream);
  (void)hipMemsetAsync(xc,    0, (size_t)N * 192 * sizeof(float), stream);
  (void)hipMemsetAsync(t3,    0, (size_t)N * 64  * sizeof(float), stream);
  (void)hipMemsetAsync(t4,    0, (size_t)N * 64  * sizeof(float), stream);
  (void)hipMemsetAsync(stats, 0, 12 * 128 * sizeof(float), stream);
  (void)hipMemsetAsync(d_out, 0, (size_t)out_size * sizeof(float), stream);

  float* S = stats;
  // conv0: pos(3) -> x(64)
  run_conv(3,   pos,  src, dst, P[0], xbuf, 64, 0,   64, h, S + 0 * 256,  S + 0 * 256 + 128,  E, stream);
  // conv1 -> xc[:,0:64]
  run_conv(64,  xbuf, src, dst, P[1], xc,  192, 0,   64, h, S + 1 * 256,  S + 1 * 256 + 128,  E, stream);
  // conv2 -> xc[:,64:128], then square
  run_conv(64,  xbuf, src, dst, P[2], xc,  192, 64,  64, h, S + 2 * 256,  S + 2 * 256 + 128,  E, stream);
  square_cols<<<(N * 16 + 255) / 256, 256, 0, stream>>>((float4*)xc, N * 16);
  // conv3 -> t3, conv4 -> t4, xc[:,128:192] = t3^2 + t4^2
  run_conv(64,  xbuf, src, dst, P[3], t3,   64, 0,   64, h, S + 3 * 256,  S + 3 * 256 + 128,  E, stream);
  run_conv(64,  xbuf, src, dst, P[4], t4,   64, 0,   64, h, S + 4 * 256,  S + 4 * 256 + 128,  E, stream);
  combine4<<<(N * 16 + 255) / 256, 256, 0, stream>>>((const float4*)t3, (const float4*)t4, (float4*)xc, N * 16);
  // conv5: xc(192) -> out(3)
  run_conv(192, xc,   src, dst, P[5], (float*)d_out, 3, 0, 3, h, S + 5 * 256, S + 5 * 256 + 128, E, stream);
}
